// Attention_17721035063654
// MI455X (gfx1250) — matrix operand order repaired and checked by reading, not compiled, not run on hardware
//
#include <hip/hip_runtime.h>
#include <cstdint>

// ---------------------------------------------------------------------------
// CDNA5 (gfx1250) attention: bf16 WMMA pipeline, f32 accumulate.
// Staging: GLOBAL_LOAD_ASYNC_TO_LDS_B128 for contiguous tiles (ASYNCcnt),
// transposed LDS layouts so every WMMA fragment read is one b128/b256 load.
// ---------------------------------------------------------------------------

typedef __attribute__((ext_vector_type(16))) __bf16 bf16x16;
typedef __attribute__((ext_vector_type(8)))  __bf16 bf16x8;
typedef __attribute__((ext_vector_type(8)))  float  f32x8;

#define WMMA_BF16(a, b, c) \
  __builtin_amdgcn_wmma_f32_16x16x32_bf16(false, (a), false, (b), (short)0, (c), false, false)

#define S_LEN   2048
#define HID     2048
#define NHEADS  32
#define NKV     8
#define HDIM    64

// LDS byte offset of a generic pointer to __shared__ (generic LDS addr keeps
// the LDS offset in addr[31:0] per the CDNA5 aperture scheme).
static __device__ __forceinline__ uint32_t lds_off(const void* p) {
  return (uint32_t)(uintptr_t)p;
}

// Async DMA: 16 bytes global -> LDS per lane, tracked by ASYNCcnt.
static __device__ __forceinline__ void async_g2l_b128(uint32_t lds_byte_off,
                                                      const void* gaddr) {
  asm volatile("global_load_async_to_lds_b128 %0, %1, off"
               :: "v"(lds_byte_off), "v"((uint64_t)(uintptr_t)gaddr)
               : "memory");
}
static __device__ __forceinline__ void wait_async0() {
  asm volatile("s_wait_asynccnt 0" ::: "memory");
}

// ------------------------- fp32 -> bf16 convert ----------------------------
__global__ __launch_bounds__(256) void f2bf_kernel(const float* __restrict__ x,
                                                   __bf16* __restrict__ y, int n) {
  for (int i = blockIdx.x * 256 + threadIdx.x; i < n; i += gridDim.x * 256)
    y[i] = (__bf16)x[i];
}

// ------------------------- bf16 WMMA GEMM ----------------------------------
// C[M,N] (f32) = A[M,K] (bf16, row major) * B[K,N] (bf16, row major)
// block tile 64x64, 8 waves: wave (wr,wc) -> rows wr*16, cols wc*32 (2 n-tiles)
// A staged row-major via async b128; B staged TRANSPOSED ([col][k]) so the
// B fragment (N fixed, K contiguous) is a single 32B LDS vector load.
__global__ __launch_bounds__(256) void gemm_bf16(const __bf16* __restrict__ A,
                                                 const __bf16* __restrict__ B,
                                                 float* __restrict__ C,
                                                 int N, int K) {
  __shared__ alignas(64) __bf16 lA[64 * 32];    // [row][k]
  __shared__ alignas(64) __bf16 lBt[64 * 32];   // [col][k]  (transposed)

  const int tid  = threadIdx.x;
  const int lane = tid & 31;
  const int wave = tid >> 5;
  const int wr   = wave & 3;
  const int wc   = wave >> 2;
  const int hl   = lane & 15;
  const int hi   = lane >> 4;

  const size_t rowBlock = (size_t)blockIdx.y * 64;
  const size_t colBlock = (size_t)blockIdx.x * 64;

  const uint32_t lA_off = lds_off(lA);

  f32x8 acc0 = {};
  f32x8 acc1 = {};

  for (int k0 = 0; k0 < K; k0 += 32) {
    // ---- A tile 64x32: one async b128 per thread (contiguous rows) --------
    {
      const int r = tid >> 2;              // 0..63
      const int c = tid & 3;               // 16B chunk within the 64B row
      const __bf16* ga = A + (rowBlock + r) * K + k0 + c * 8;
      async_g2l_b128(lA_off + (uint32_t)tid * 16, ga);
    }
    // ---- B tile 32x64 -> transposed LDS [col][k] --------------------------
    {
      const uint32_t* gB = (const uint32_t*)(B + (size_t)k0 * N + colBlock);
#pragma unroll
      for (int i = 0; i < 4; ++i) {
        const int idx   = tid + i * 256;   // 1024 dwords
        const int k     = idx >> 5;        // 0..31
        const int cpair = idx & 31;        // 2 cols per dword
        union { uint32_t u; __bf16 h[2]; } cv;
        cv.u = gB[(size_t)k * (N >> 1) + cpair];
        lBt[(2 * cpair) * 32 + k]     = cv.h[0];
        lBt[(2 * cpair + 1) * 32 + k] = cv.h[1];
      }
    }
    if (k0 + 32 < K)                        // -> global_prefetch_b8
      __builtin_prefetch((const void*)(A + rowBlock * K + k0 + 32), 0, 0);
    wait_async0();
    __syncthreads();

    // A fragment 16x32: lane<16 -> K {0..15}; lane>=16 -> K {16..31}
    bf16x16 af;
    {
      const __bf16* ar = lA + (wr * 16 + hl) * 32 + hi * 16;
      bf16x8 lo  = *(const bf16x8*)(ar);
      bf16x8 hi8 = *(const bf16x8*)(ar + 8);
#pragma unroll
      for (int j = 0; j < 8; ++j) { af[j] = lo[j]; af[8 + j] = hi8[j]; }
    }
    // B fragments: contiguous 32B vector loads from transposed LDS
    const __bf16* bb = lBt + (wc * 32 + hl) * 32 + hi * 16;
    const bf16x16 bf0 = *(const bf16x16*)(bb);
    const bf16x16 bf1 = *(const bf16x16*)(bb + 16 * 32);

    acc0 = WMMA_BF16(af, bf0, acc0);
    acc1 = WMMA_BF16(af, bf1, acc1);
    __syncthreads();
  }

  // C layout: vgpr r, lane<16 -> M=r ; lane>=16 -> M=8+r ; N = lane%16
  const size_t row = rowBlock + wr * 16 + hi * 8;
  const size_t col = colBlock + wc * 32 + hl;
#pragma unroll
  for (int r = 0; r < 8; ++r) {
    C[(row + r) * N + col]      = acc0[r];
    C[(row + r) * N + col + 16] = acc1[r];
  }
}

// ------------------------- RoPE + head-major bf16 --------------------------
// X: [S, nHeads*64] f32 -> Y: [nHeads, S, 64] bf16 (RoPE applied, * outScale)
__global__ __launch_bounds__(256) void rope_kernel(const float* __restrict__ X,
                                                   __bf16* __restrict__ Y,
                                                   int nHeads, float outScale) {
  const int idx = blockIdx.x * 256 + threadIdx.x;     // over nHeads*S*32 pairs
  if (idx >= nHeads * S_LEN * 32) return;
  const int i = idx & 31;                 // pair index within head dim
  const int s = (idx >> 5) & (S_LEN - 1); // sequence position
  const int h = idx >> 16;                // 32*2048 = 65536 pairs per head

  const float inv = __powf(10000.0f, -((float)(2 * i) * (1.0f / 64.0f)));
  const float ang = (float)s * inv;
  float sn, cs;
  __sincosf(ang, &sn, &cs);

  const float* xp = X + (size_t)s * (nHeads * 64) + h * 64 + 2 * i;
  const float e = xp[0], o = xp[1];
  __bf16* yp = Y + ((size_t)h * S_LEN + s) * 64 + 2 * i;
  yp[0] = (__bf16)((e * cs - o * sn) * outScale);
  yp[1] = (__bf16)((e * sn + o * cs) * outScale);
}

// V: [S, NKV*64] f32 -> [NKV, S, 64] bf16
__global__ __launch_bounds__(256) void vconv_kernel(const float* __restrict__ Vf,
                                                    __bf16* __restrict__ Vb) {
  const int idx = blockIdx.x * 256 + threadIdx.x;
  if (idx >= NKV * S_LEN * HDIM) return;
  const int d = idx & 63;
  const int s = (idx >> 6) & (S_LEN - 1);
  const int h = idx >> 17;
  Vb[((size_t)h * S_LEN + s) * 64 + d] = (__bf16)Vf[(size_t)s * (NKV * 64) + h * 64 + d];
}

// ------------------------- flash attention (non-causal) --------------------
// grid: (S/128, NHEADS). 8 waves, wave w owns 16 query rows. key blocks of 32.
// K staged row-major (async b128) -> K^T fragments are contiguous.
// V staged TRANSPOSED ([d][key]) -> V fragments are contiguous.
__global__ __launch_bounds__(256) void attn_kernel(const __bf16* __restrict__ Qb,
                                                   const __bf16* __restrict__ Kb,
                                                   const __bf16* __restrict__ Vb,
                                                   __bf16* __restrict__ AOb) {
  __shared__ alignas(64) __bf16 sK[32 * 64];       // [key][d]
  __shared__ alignas(64) __bf16 sVt[64 * 32];      // [d][key]  (transposed)
  __shared__ alignas(64) __bf16 sP[8][16 * 32];    // per-wave P transpose scratch

  const int h    = blockIdx.y;          // query head
  const int hk   = h >> 2;              // GQA: kv head = h / 4
  const int tid  = threadIdx.x;
  const int lane = tid & 31;
  const int wave = tid >> 5;
  const int hl   = lane & 15;
  const int hi   = lane >> 4;
  const int row0 = blockIdx.x * 128 + wave * 16;

  const uint32_t sK_off = lds_off(sK);

  // Q fragments for d 0..31 and 32..63 (A layout), loaded once, vectorized
  bf16x16 qf[2];
#pragma unroll
  for (int dt = 0; dt < 2; ++dt) {
    const __bf16* qr = Qb + ((size_t)h * S_LEN + row0 + hl) * 64 + dt * 32 + hi * 16;
    bf16x8 lo  = *(const bf16x8*)(qr);
    bf16x8 hi8 = *(const bf16x8*)(qr + 8);
#pragma unroll
    for (int j = 0; j < 8; ++j) { qf[dt][j] = lo[j]; qf[dt][8 + j] = hi8[j]; }
  }

  float m[8], l[8];
  f32x8 o[4] = {};
#pragma unroll
  for (int r = 0; r < 8; ++r) { m[r] = -1e30f; l[r] = 0.0f; }

  for (int kb0 = 0; kb0 < S_LEN; kb0 += 32) {
    // ---- K block 32x64: contiguous -> one async b128 per thread ----------
    async_g2l_b128(sK_off + (uint32_t)tid * 16,
                   Kb + ((size_t)hk * S_LEN + kb0) * 64 + (size_t)tid * 8);
    // ---- V block 32x64 -> transposed LDS [d][key] ------------------------
    {
      const uint32_t* gV = (const uint32_t*)(Vb + ((size_t)hk * S_LEN + kb0) * 64);
#pragma unroll
      for (int i = 0; i < 4; ++i) {
        const int idx   = tid + i * 256;  // 1024 dwords, contiguous block
        const int key   = idx >> 5;       // 0..31
        const int dpair = idx & 31;       // 2 d-values per dword
        union { uint32_t u; __bf16 hh[2]; } cv;
        cv.u = gV[idx];
        sVt[(2 * dpair) * 32 + key]     = cv.hh[0];
        sVt[(2 * dpair + 1) * 32 + key] = cv.hh[1];
      }
    }
    wait_async0();
    __syncthreads();

    // S = Q * K^T : two 16x16 tiles over 32 keys, contraction over d (2 steps)
    f32x8 s0 = {}, s1 = {};
#pragma unroll
    for (int dt = 0; dt < 2; ++dt) {
      const bf16x16 bk0 = *(const bf16x16*)(sK + (0  + hl) * 64 + dt * 32 + hi * 16);
      const bf16x16 bk1 = *(const bf16x16*)(sK + (16 + hl) * 64 + dt * 32 + hi * 16);
      s0 = WMMA_BF16(qf[dt], bk0, s0);
      s1 = WMMA_BF16(qf[dt], bk1, s1);
    }

    // online softmax (rows 0-7 in lanes 0-15, rows 8-15 in lanes 16-31;
    // xor masks 1..8 stay within each 16-lane half)
    float p0[8], p1[8], sc[8];
#pragma unroll
    for (int r = 0; r < 8; ++r) {
      float v = fmaxf(s0[r], s1[r]);
      v = fmaxf(v, __shfl_xor(v, 1));
      v = fmaxf(v, __shfl_xor(v, 2));
      v = fmaxf(v, __shfl_xor(v, 4));
      v = fmaxf(v, __shfl_xor(v, 8));
      const float mn = fmaxf(m[r], v);
      sc[r] = __expf(m[r] - mn);
      m[r]  = mn;
      p0[r] = __expf(s0[r] - mn);
      p1[r] = __expf(s1[r] - mn);
      float rs = p0[r] + p1[r];
      rs += __shfl_xor(rs, 1);
      rs += __shfl_xor(rs, 2);
      rs += __shfl_xor(rs, 4);
      rs += __shfl_xor(rs, 8);
      l[r] = l[r] * sc[r] + rs;
    }
#pragma unroll
    for (int t = 0; t < 4; ++t)
#pragma unroll
      for (int r = 0; r < 8; ++r) o[t][r] *= sc[r];

    // transpose P (C layout) -> A fragment layout via wave-private LDS
    __bf16* pp = sP[wave];
#pragma unroll
    for (int r = 0; r < 8; ++r) {
      pp[(hi * 8 + r) * 32 + hl]      = (__bf16)p0[r];
      pp[(hi * 8 + r) * 32 + 16 + hl] = (__bf16)p1[r];
    }
    bf16x16 pf;   // LDS ops are in-order within a wave
    {
      bf16x8 lo  = *(const bf16x8*)(pp + hl * 32 + hi * 16);
      bf16x8 hi8 = *(const bf16x8*)(pp + hl * 32 + 8 + hi * 16);
#pragma unroll
      for (int j = 0; j < 8; ++j) { pf[j] = lo[j]; pf[8 + j] = hi8[j]; }
    }

    // O += P * V : 4 d-tiles, each B fragment one contiguous 32B LDS load
#pragma unroll
    for (int t = 0; t < 4; ++t) {
      const bf16x16 vf = *(const bf16x16*)(sVt + (t * 16 + hl) * 32 + hi * 16);
      o[t] = WMMA_BF16(pf, vf, o[t]);
    }
    __syncthreads();
  }

  // epilogue: normalize and write attn_out [S, NHEADS*64] bf16
#pragma unroll
  for (int t = 0; t < 4; ++t)
#pragma unroll
    for (int r = 0; r < 8; ++r) {
      const size_t row = row0 + hi * 8 + r;
      const size_t col = (size_t)h * 64 + t * 16 + hl;
      AOb[row * HID + col] = (__bf16)(o[t][r] / l[r]);
    }
}

// ---------------------------------------------------------------------------
extern "C" void kernel_launch(void* const* d_in, const int* in_sizes, int n_in,
                              void* d_out, int out_size, void* d_ws, size_t ws_size,
                              hipStream_t stream) {
  (void)in_sizes; (void)n_in; (void)out_size; (void)ws_size;
  const float* H  = (const float*)d_in[0];
  const float* Wq = (const float*)d_in[1];
  const float* Wk = (const float*)d_in[2];
  const float* Wv = (const float*)d_in[3];
  const float* Wo = (const float*)d_in[4];
  float* out = (float*)d_out;

  char* ws = (char*)d_ws;
  auto alloc = [&](size_t bytes) -> char* {
    char* p = ws;
    ws += (bytes + 255) & ~(size_t)255;
    return p;
  };

  __bf16* Hb  = (__bf16*)alloc((size_t)S_LEN * HID * 2);          // 8 MB
  __bf16* Wqb = (__bf16*)alloc((size_t)HID * HID * 2);            // 8 MB
  __bf16* Wkb = (__bf16*)alloc((size_t)HID * (NKV * 64) * 2);     // 2 MB
  __bf16* Wvb = (__bf16*)alloc((size_t)HID * (NKV * 64) * 2);     // 2 MB
  __bf16* Wob = (__bf16*)alloc((size_t)HID * HID * 2);            // 8 MB
  float*  Qf  = (float*)alloc((size_t)S_LEN * HID * 4);           // 16 MB
  float*  Kf  = (float*)alloc((size_t)S_LEN * (NKV * 64) * 4);    // 4 MB
  float*  Vf  = (float*)alloc((size_t)S_LEN * (NKV * 64) * 4);    // 4 MB
  __bf16* Qbh = (__bf16*)alloc((size_t)NHEADS * S_LEN * 64 * 2);  // 8 MB
  __bf16* Kbh = (__bf16*)alloc((size_t)NKV * S_LEN * 64 * 2);     // 2 MB
  __bf16* Vbh = (__bf16*)alloc((size_t)NKV * S_LEN * 64 * 2);     // 2 MB
  __bf16* AOb = (__bf16*)alloc((size_t)S_LEN * HID * 2);          // 8 MB

  // 1) convert inputs to bf16
  f2bf_kernel<<<4096, 256, 0, stream>>>(H,  Hb,  S_LEN * HID);
  f2bf_kernel<<<4096, 256, 0, stream>>>(Wq, Wqb, HID * HID);
  f2bf_kernel<<<1024, 256, 0, stream>>>(Wk, Wkb, HID * NKV * 64);
  f2bf_kernel<<<1024, 256, 0, stream>>>(Wv, Wvb, HID * NKV * 64);
  f2bf_kernel<<<4096, 256, 0, stream>>>(Wo, Wob, HID * HID);

  // 2) QKV projections (WMMA bf16, f32 out)
  gemm_bf16<<<dim3(HID / 64, S_LEN / 64), 256, 0, stream>>>(Hb, Wqb, Qf, HID, HID);
  gemm_bf16<<<dim3((NKV * 64) / 64, S_LEN / 64), 256, 0, stream>>>(Hb, Wkb, Kf, NKV * 64, HID);
  gemm_bf16<<<dim3((NKV * 64) / 64, S_LEN / 64), 256, 0, stream>>>(Hb, Wvb, Vf, NKV * 64, HID);

  // 3) RoPE (+ fold softmax 1/sqrt(64) into Q) and head-major relayout
  rope_kernel<<<(NHEADS * S_LEN * 32) / 256, 256, 0, stream>>>(Qf, Qbh, NHEADS, 0.125f);
  rope_kernel<<<(NKV * S_LEN * 32) / 256, 256, 0, stream>>>(Kf, Kbh, NKV, 1.0f);
  vconv_kernel<<<(NKV * S_LEN * HDIM) / 256, 256, 0, stream>>>(Vf, Vbh);

  // 4) attention (flash-style, WMMA for QK^T and PV)
  attn_kernel<<<dim3(S_LEN / 128, NHEADS), 256, 0, stream>>>(Qbh, Kbh, Vbh, AOb);

  // 5) output projection -> f32 d_out
  gemm_bf16<<<dim3(HID / 64, S_LEN / 64), 256, 0, stream>>>(AOb, Wob, out, HID, HID);
}
